// CNN2D_37495064494620
// MI455X (gfx1250) — matrix operand order repaired and checked by reading, not compiled, not run on hardware
//
#include <hip/hip_runtime.h>
#include <hip/hip_bf16.h>

// CDNA5 (gfx1250) lattice-CNN log-psi.
//
// One wave32 = 16 samples. The 3x3 wrap-conv (1->4ch) runs on the WMMA pipe,
// with THREE output pixels packed per V_WMMA_F32_16X16X32_F16:
//   K = slot*9 + tap   (slot 0..2 = pixel within batch, tap 0..8)
//   N = slot*4 + f     (f = filter 0..3)  -> B[K,N] = w[f][tap] iff K-window
//   M = sample 0..15
// 25 pool-groups are processed as 9 batches of 3; the 4 pool members (d=0..3)
// of a batch are 4 WMMAs sharing one B fragment, so the 2x2 maxpool is a
// per-lane v_max over the 4 C fragments (lane encodes slot*4+f, VGPR index
// encodes sample). 36 WMMAs/wave (21% useful occupancy) keeps the matrix pipe
// below the 23.3 TB/s HBM floor.
//
// A-matrix layout per ISA 7.12.2 (16-bit A 16x32): lane l -> row M=l%16;
// lanes<16 hold K = e (e<8) / e+8 (e>=8); lanes>=16 hold K = e+8 / e+16.
// B-matrix (32x16): element k of lane l is row K = k (k<8) / k+8 (k>=8), plus 8 for l>=16 (same as A),
// column N = l%16.
//
// Pooled tiles go to LDS; deconv(2x2,s=2) + log|z| + sign-parity reads them
// back one sample per lane-pair and reduces with a single shfl_xor(16).

typedef __attribute__((ext_vector_type(16))) _Float16 v16h;
typedef __attribute__((ext_vector_type(8)))  float    v8f;

#define WAVES_PER_BLOCK 8
#define THREADS_PER_BLOCK (WAVES_PER_BLOCK * 32)

__global__ __launch_bounds__(THREADS_PER_BLOCK)
void nqs_logpsi_kernel(const float* __restrict__ v,
                       const float* __restrict__ conv_w,    // [4,1,3,3] = 36
                       const float* __restrict__ conv_b,    // [4]
                       const float* __restrict__ deconv_w,  // [4,1,2,2] = 16
                       const float* __restrict__ deconv_b,  // [1]
                       float* __restrict__ out_sign,        // [nsamp]
                       float* __restrict__ out_log,         // [nsamp]
                       int nsamp)
{
    // pooled_lds[wave][g = f*25 + (i*5+j)][sample 0..15]
    __shared__ __align__(16) float pooled_lds[WAVES_PER_BLOCK][100][16];

    if ((long)blockIdx.x * (WAVES_PER_BLOCK * 16) >= (long)nsamp) return; // block-uniform

    const int tid  = threadIdx.x;
    const int wib  = tid >> 5;
    const int lane = tid & 31;
    const int wave = blockIdx.x * WAVES_PER_BLOCK + wib;
    const int s0   = wave * 16;                 // first sample of this wave
    const int m    = lane & 15;                 // row (sample) / column index role
    const bool lo  = lane < 16;
    const int sm   = s0 + m;                    // this lane's sample
    const int smL  = sm < nsamp ? sm : (nsamp - 1);

    // ---------------- load this lane's sample (100 f32) -> f16 registers ----
    _Float16 xh[100];
    const float4* xp = reinterpret_cast<const float4*>(v + (long)smL * 100);
    #pragma unroll
    for (int q = 0; q < 25; ++q) {
        float4 t4 = xp[q];
        xh[4*q + 0] = (_Float16)t4.x;
        xh[4*q + 1] = (_Float16)t4.y;
        xh[4*q + 2] = (_Float16)t4.z;
        xh[4*q + 3] = (_Float16)t4.w;
    }

    const _Float16 h0 = (_Float16)0.f;
    const int slot = m >> 2;                    // pixel slot this column serves
    const int fc   = m & 3;                     // filter this column serves

    // ---------------- B fragment: w[f][tap] shifted into K-window 9*slot ----
    v16h bfrag;
    #pragma unroll
    for (int k = 0; k < 16; ++k) {
        const int  K    = ((k < 8) ? k : (k + 8)) + (lo ? 0 : 8);
        const int  rel  = K - 9 * slot;
        const bool cond = (m < 12) && (rel >= 0) && (rel < 9);
        const int  addr = cond ? (fc * 9 + rel) : 0;
        const float wv  = conv_w[addr];
        bfrag[k] = cond ? (_Float16)wv : h0;
    }
    const float bias = conv_b[fc];

    const v8f czero = {0.f, 0.f, 0.f, 0.f, 0.f, 0.f, 0.f, 0.f};

    // ---------------- conv + bias + maxpool via WMMA (9 batches x 4) --------
    #pragma unroll
    for (int b = 0; b < 9; ++b) {               // batch of 3 pool-groups
        v8f pmax = czero;
        #pragma unroll
        for (int d = 0; d < 4; ++d) {           // 2x2 pool member
            v16h a;
            #pragma unroll
            for (int e = 0; e < 16; ++e) {
                const int Klo = (e < 8) ? e : (e + 8);   // K for lanes<16
                const int Khi = Klo + 8;                 // K for lanes>=16
                _Float16 tlo = h0, thi = h0;
                // tap value for a given K (compile-time constant indices)
                {
                    const int K = Klo;                   // 0..23, valid K<27
                    if (K < 27) {
                        const int s = K / 9, t = K % 9;
                        const int G = 3*b + s;           // pool-group (i*5+j)
                        const int r = 2*(G/5) + (d >> 1);
                        const int c = 2*(G%5) + (d & 1);
                        const int rr = (r + t/3 + 9) % 10;
                        const int cc = (c + t%3 + 9) % 10;
                        tlo = xh[rr*10 + cc];
                    }
                }
                {
                    const int K = Khi;                   // 8..31, valid K<27
                    if (K < 27) {
                        const int s = K / 9, t = K % 9;
                        const int G = 3*b + s;
                        const int r = 2*(G/5) + (d >> 1);
                        const int c = 2*(G%5) + (d & 1);
                        const int rr = (r + t/3 + 9) % 10;
                        const int cc = (c + t%3 + 9) % 10;
                        thi = xh[rr*10 + cc];
                    }
                }
                a[e] = lo ? tlo : thi;
            }

            v8f cd = __builtin_amdgcn_wmma_f32_16x16x32_f16(
                /*neg_a=*/false, a, /*neg_b=*/false, bfrag,
                /*c_mod=*/(short)0, czero, /*reuse_a=*/false, /*reuse_b=*/false);

            if (d == 0) {
                pmax = cd;
            } else {
                #pragma unroll
                for (int e = 0; e < 8; ++e) pmax[e] = fmaxf(pmax[e], cd[e]);
            }
        }
        // store pooled values: lane column m<12 holds (slot, f) of this batch
        const int G = 3*b + slot;
        if (m < 12 && G < 25) {
            const int g = fc * 25 + G;
            float* dst = &pooled_lds[wib][g][lo ? 0 : 8];
            float4 a4 = { pmax[0] + bias, pmax[1] + bias, pmax[2] + bias, pmax[3] + bias };
            float4 b4 = { pmax[4] + bias, pmax[5] + bias, pmax[6] + bias, pmax[7] + bias };
            *reinterpret_cast<float4*>(dst)     = a4;
            *reinterpret_cast<float4*>(dst + 4) = b4;
        }
    }
    __syncthreads();

    // ---------------- deconv(2x2, s=2) + log|z| + sign parity ---------------
    // Uniform pointer + constant indices -> scalar broadcasts.
    float dw[16];
    #pragma unroll
    for (int k = 0; k < 16; ++k) dw[k] = deconv_w[k];
    const float db = deconv_b[0];

    const int h = lane >> 4;                       // lane pair splits the 25 blocks
    const float* pb = &pooled_lds[wib][0][m];      // sample m, stride 16 floats per g
    float acc = 0.f;
    unsigned sgn = 0u;
    #pragma unroll
    for (int q = 0; q < 13; ++q) {
        const int blk = 2*q + h;                   // pooled position i*5+j
        if (blk < 25) {
            const float p0 = pb[(0*25 + blk) * 16];
            const float p1 = pb[(1*25 + blk) * 16];
            const float p2 = pb[(2*25 + blk) * 16];
            const float p3 = pb[(3*25 + blk) * 16];
            #pragma unroll
            for (int kl = 0; kl < 4; ++kl) {
                const float z = db + p0*dw[kl] + p1*dw[4+kl] + p2*dw[8+kl] + p3*dw[12+kl];
                sgn ^= __float_as_uint(z) & 0x80000000u;   // sign-bit parity
                acc += __log2f(fabsf(z));                  // v_log_f32; scale by ln2 once
            }
        }
    }
    // combine the two half-lanes handling the same sample
    acc += __shfl_xor(acc, 16, 32);
    sgn ^= (unsigned)__shfl_xor((int)sgn, 16, 32);

    if (lo && sm < nsamp) {
        out_sign[sm] = (sgn == 0u) ? 1.0f : -1.0f;
        out_log[sm]  = acc * 0.69314718055994530942f;      // ln2 * sum(log2|z|)
    }
}

extern "C" void kernel_launch(void* const* d_in, const int* in_sizes, int n_in,
                              void* d_out, int out_size, void* d_ws, size_t ws_size,
                              hipStream_t stream) {
    const float* v        = (const float*)d_in[0];
    const float* conv_w   = (const float*)d_in[1];
    const float* conv_b   = (const float*)d_in[2];
    const float* deconv_w = (const float*)d_in[3];
    const float* deconv_b = (const float*)d_in[4];
    // d_in[5] ("data") is unused by the reference computation.

    const int nsamp = in_sizes[0] / 100;           // B = 262144
    float* out_sign = (float*)d_out;               // tuple order: (sign_prod, log_sum)
    float* out_log  = out_sign + nsamp;

    const int samples_per_block = WAVES_PER_BLOCK * 16;   // 128
    const int blocks = (nsamp + samples_per_block - 1) / samples_per_block;

    nqs_logpsi_kernel<<<blocks, THREADS_PER_BLOCK, 0, stream>>>(
        v, conv_w, conv_b, deconv_w, deconv_b, out_sign, out_log, nsamp);
}
